// MultiHeadAttention_29996051595713
// MI455X (gfx1250) — matrix operand order repaired and checked by reading, not compiled, not run on hardware
//
#include <hip/hip_runtime.h>
#include <hip/hip_fp16.h>

typedef __attribute__((ext_vector_type(16))) _Float16 v16h;
typedef __attribute__((ext_vector_type(8)))  _Float16 v8h;
typedef __attribute__((ext_vector_type(8)))  float    v8f;
typedef __attribute__((ext_vector_type(4)))  float    v4f;

#define HEADS 12
#define EMB   768
#define DKDIM 64
#define SEQ   4096
#define BATCH 2
#define MROWS (BATCH * SEQ)  // 8192

// --------------------------------------------------------------------------
// CDNA5 async global->LDS copy (ASYNCcnt) with compile-safe fallback.
// Builtin signature (from hipcc diagnostic): param 1 is a pointer to a
// 16-byte int vector in the global address space; param 2 the LDS-side one.
// --------------------------------------------------------------------------
#if __has_builtin(__builtin_amdgcn_global_load_async_to_lds_b128)
#define HAS_ASYNC_LDS 1
typedef int v4i_vs __attribute__((vector_size(16)));
typedef __attribute__((address_space(1))) v4i_vs* gas_v4i_ptr;
typedef __attribute__((address_space(3))) v4i_vs* las_v4i_ptr;
#endif

#if __has_builtin(__builtin_amdgcn_s_wait_asynccnt)
#define WAIT_ASYNC(n) __builtin_amdgcn_s_wait_asynccnt(n)
#else
#define WAIT_ASYNC(n) asm volatile("s_wait_asynccnt %0" ::"i"(n) : "memory")
#endif

// ---------------------------------------------------------------------------
// WMMA fragment helpers (wave32, 16x16x32 f16 -> f32)
// A fragment (16x32, MxK):  lane&15 = M row; lane>>4 selects K sub-runs:
//   elements 0..7  -> K = half*8 + e ; elements 8..15 -> K = 16 + half*8 + e-8
// B fragment (32x16, KxN):  lane&15 = N col; same K sub-runs as A fragment
// C/D fragment (16x16 f32): element r -> M = r + 8*half, N = lane&15
// ---------------------------------------------------------------------------
__device__ __forceinline__ v8f wmma_f16(v16h a, v16h b, v8f c) {
  return __builtin_amdgcn_wmma_f32_16x16x32_f16(
      /*neg_a=*/false, a, /*neg_b=*/false, b,
      /*c_mod=*/(short)0, c, /*reuse_a=*/false, /*reuse_b=*/false);
}

// A fragment from row-major f32 matrix (convert to f16 in regs)
__device__ __forceinline__ v16h load_a_f32(const float* X, int ld, int row0,
                                           int k0, int lane) {
  const int half = lane >> 4, m = lane & 15;
  const float* p = X + (size_t)(row0 + m) * ld + k0 + half * 8;
  v4f f0 = *(const v4f*)(p);
  v4f f1 = *(const v4f*)(p + 4);
  v4f f2 = *(const v4f*)(p + 16);
  v4f f3 = *(const v4f*)(p + 20);
  v16h a;
#pragma unroll
  for (int i = 0; i < 4; ++i) {
    a[i]      = (_Float16)f0[i];
    a[4 + i]  = (_Float16)f1[i];
    a[8 + i]  = (_Float16)f2[i];
    a[12 + i] = (_Float16)f3[i];
  }
  return a;
}

// A fragment from row-major f16 matrix (two 16B loads per lane)
__device__ __forceinline__ v16h load_a_f16(const _Float16* X, int ld, int row0,
                                           int k0, int lane) {
  const int half = lane >> 4, m = lane & 15;
  const _Float16* p = X + (size_t)(row0 + m) * ld + k0 + half * 8;
  v8h lo = *(const v8h*)(p);
  v8h hi = *(const v8h*)(p + 16);
  v16h a;
#pragma unroll
  for (int i = 0; i < 8; ++i) { a[i] = lo[i]; a[8 + i] = hi[i]; }
  return a;
}

// B fragment from N-major f16 matrix Bt[N][K] (two 16B loads per lane)
__device__ __forceinline__ v16h load_b_f16(const _Float16* Bt, int ldb, int n0,
                                           int k0, int lane) {
  const int half = lane >> 4, n = lane & 15;
  const _Float16* p = Bt + (size_t)(n0 + n) * ldb + k0 + half * 8;
  v8h lo = *(const v8h*)(p);
  v8h hi = *(const v8h*)(p + 16);
  v16h b;
#pragma unroll
  for (int i = 0; i < 8; ++i) { b[i] = lo[i]; b[8 + i] = hi[i]; }
  return b;
}

// ---------------------------------------------------------------------------
// Kernel 1: transpose + convert 4 weight matrices [in][out]f32 -> [out][in]f16
// ---------------------------------------------------------------------------
__global__ void cvt_weights(const float* __restrict__ Wq,
                            const float* __restrict__ Wk,
                            const float* __restrict__ Wv,
                            const float* __restrict__ Wf,
                            _Float16* __restrict__ out) {
  int idx = blockIdx.x * blockDim.x + threadIdx.x;  // 4*768*768 threads
  int w = idx / (EMB * EMB);
  int r = idx % (EMB * EMB);
  int n = r / EMB, k = r % EMB;
  const float* W = (w == 0) ? Wq : (w == 1) ? Wk : (w == 2) ? Wv : Wf;
  out[(size_t)w * EMB * EMB + (size_t)n * EMB + k] =
      (_Float16)W[(size_t)k * EMB + n];
}

// ---------------------------------------------------------------------------
// Kernel 2: QKV projections.  Each wave: 16x64 output tile, K loop over 768.
// Q,K stored [B,H,S,64] row-major f16; V stored transposed [B,H,64,S] f16.
// ---------------------------------------------------------------------------
__global__ __launch_bounds__(128) void qkv_proj(
    const float* __restrict__ Xq, const float* __restrict__ Xk,
    const float* __restrict__ Xv, const _Float16* __restrict__ Wt,
    const float* __restrict__ bq, const float* __restrict__ bk,
    const float* __restrict__ bv, _Float16* __restrict__ Qh,
    _Float16* __restrict__ Kh, _Float16* __restrict__ Vt) {
  const int lane = threadIdx.x & 31;
  const int wave = threadIdx.x >> 5;
  const int which = blockIdx.z;
  const float* X = (which == 0) ? Xq : (which == 1) ? Xk : Xv;
  const _Float16* W = Wt + (size_t)which * EMB * EMB;
  const float* bias = (which == 0) ? bq : (which == 1) ? bk : bv;

  const int row0 = blockIdx.x * 64 + wave * 16;
  const int n0 = blockIdx.y * 64;

  v8f acc[4] = {};
  for (int k0 = 0; k0 < EMB; k0 += 32) {
    v16h a = load_a_f32(X, EMB, row0, k0, lane);
#pragma unroll
    for (int nt = 0; nt < 4; ++nt) {
      v16h bfrag = load_b_f16(W, EMB, n0 + nt * 16, k0, lane);
      acc[nt] = wmma_f16(a, bfrag, acc[nt]);
    }
  }

  const int half = lane >> 4, nl = lane & 15;
#pragma unroll
  for (int nt = 0; nt < 4; ++nt) {
    const int gc = n0 + nt * 16 + nl;
    const float bb = bias[gc];
    const int h = gc >> 6, d = gc & 63;
#pragma unroll
    for (int r = 0; r < 8; ++r) {
      const int gr = row0 + r + 8 * half;
      const int b = gr >> 12;      // / SEQ
      const int s = gr & (SEQ - 1);
      const float val = acc[nt][r] + bb;
      const size_t hb = (size_t)(b * HEADS + h);
      if (which == 2) {
        Vt[(hb * DKDIM + d) * SEQ + s] = (_Float16)val;
      } else {
        _Float16* dst = (which == 0) ? Qh : Kh;
        dst[(hb * SEQ + s) * DKDIM + d] = (_Float16)val;
      }
    }
  }
}

// ---------------------------------------------------------------------------
// Kernel 3: flash attention.
// Block = 8 waves x 16 query rows = 128 queries.  Key loop in tiles of 64.
// K/V tiles are staged into LDS with double-buffered async global->LDS DMA
// (ASYNCcnt), shared by all 8 waves (8x L2-traffic reduction).  Scores stay
// in D-fragments; online softmax via __shfl_xor; P staged f16 through LDS.
// ---------------------------------------------------------------------------
__global__ __launch_bounds__(256) void attn(
    const _Float16* __restrict__ Qh, const _Float16* __restrict__ Kh,
    const _Float16* __restrict__ Vt, const int* __restrict__ mask,
    _Float16* __restrict__ Ch) {
  // 2x8KB K, 2x8KB V, 8x2KB P = 48 KB LDS
  __shared__ __align__(32) _Float16 sK[2][64 * 64];
  __shared__ __align__(32) _Float16 sV[2][64 * 64];
  __shared__ __align__(32) _Float16 sP[8][16 * 64];

  const int tid = threadIdx.x;
  const int lane = tid & 31;
  const int wave = tid >> 5;
  const int half = lane >> 4, nl = lane & 15;
  const int h = blockIdx.y, b = blockIdx.z;
  const int q0 = blockIdx.x * 128 + wave * 16;

  const size_t bh = (size_t)(b * HEADS + h);
  const _Float16* Q = Qh + bh * SEQ * DKDIM;
  const _Float16* K = Kh + bh * SEQ * DKDIM;  // [S, 64]
  const _Float16* V = Vt + bh * DKDIM * SEQ;  // [64, S]
  const int* mk = mask + (size_t)b * SEQ;

  // cooperative 64x64 f16 tile stage: 256 threads x 2 chunks x 16B = 8 KB
  // grs = global row stride in halfs; LDS tile row stride = 64 halfs
  auto stage_tile = [&](const _Float16* g, size_t grs, _Float16* l) {
#pragma unroll
    for (int i = 0; i < 2; ++i) {
      const int chunk = tid + i * 256;     // 0..511
      const int row = chunk >> 3;          // 64 rows
      const int col = (chunk & 7) * 8;     // halfs, 16B granules
      const _Float16* gs = g + (size_t)row * grs + col;
      _Float16* ld = l + row * 64 + col;
#ifdef HAS_ASYNC_LDS
      __builtin_amdgcn_global_load_async_to_lds_b128(
          (gas_v4i_ptr)gs, (las_v4i_ptr)ld, 0, 0);
#else
      *(v8h*)ld = *(const v8h*)gs;
#endif
    }
  };

  // Q tile A-fragments are invariant over the key loop
  v16h aq0 = load_a_f16(Q, DKDIM, q0, 0, lane);
  v16h aq1 = load_a_f16(Q, DKDIM, q0, 32, lane);

  v8f ctx[4] = {};
  float mrow[8], lrow[8];
#pragma unroll
  for (int r = 0; r < 8; ++r) { mrow[r] = -__builtin_inff(); lrow[r] = 0.f; }

  const float scale = 0.125f;  // 1/sqrt(64)
  const int nblk = SEQ / 64;

  // prestage tile 0  (2 async ops for K + 2 for V per wave)
  stage_tile(K, DKDIM, sK[0]);
  stage_tile(V, SEQ, sV[0]);

  for (int it = 0; it < nblk; ++it) {
    const int cur = it & 1, nxt = cur ^ 1;
    const int key0 = it * 64;

    if (it + 1 < nblk) {  // issue next tile's DMA, then drain current's
      stage_tile(K + (size_t)(key0 + 64) * DKDIM, DKDIM, sK[nxt]);
      stage_tile(V + (key0 + 64), SEQ, sV[nxt]);
      WAIT_ASYNC(4);
    } else {
      WAIT_ASYNC(0);
    }
    __syncthreads();  // current K/V tile visible to all waves

    const _Float16* Kt = sK[cur];
    const _Float16* Vtile = sV[cur];

    // ---- scores S = Q K^T for a 16x64 tile (B-fragments from LDS) ------
    v8f sc[4] = {};
    int kmask[4];
#pragma unroll
    for (int nt = 0; nt < 4; ++nt) {
      v16h bk0 = load_b_f16(Kt, DKDIM, nt * 16, 0, lane);
      v16h bk1 = load_b_f16(Kt, DKDIM, nt * 16, 32, lane);
      sc[nt] = wmma_f16(aq0, bk0, sc[nt]);
      sc[nt] = wmma_f16(aq1, bk1, sc[nt]);
      kmask[nt] = mk[key0 + nt * 16 + nl];
    }
#pragma unroll
    for (int nt = 0; nt < 4; ++nt)
#pragma unroll
      for (int r = 0; r < 8; ++r)
        sc[nt][r] = kmask[nt] ? sc[nt][r] * scale : -1.0e9f;

    // ---- online softmax (rows live on 16-lane halves) ------------------
    float alpha[8];
#pragma unroll
    for (int r = 0; r < 8; ++r) {
      float t = fmaxf(fmaxf(sc[0][r], sc[1][r]), fmaxf(sc[2][r], sc[3][r]));
#pragma unroll
      for (int off = 1; off < 16; off <<= 1)
        t = fmaxf(t, __shfl_xor(t, off, 32));
      const float mn = fmaxf(mrow[r], t);
      alpha[r] = __expf(mrow[r] - mn);
      mrow[r] = mn;
    }
#pragma unroll
    for (int r = 0; r < 8; ++r) {
      float psum = 0.f;
#pragma unroll
      for (int nt = 0; nt < 4; ++nt) {
        const float p = __expf(sc[nt][r] - mrow[r]);
        sc[nt][r] = p;
        psum += p;
      }
#pragma unroll
      for (int off = 1; off < 16; off <<= 1)
        psum += __shfl_xor(psum, off, 32);
      lrow[r] = lrow[r] * alpha[r] + psum;
#pragma unroll
      for (int nt = 0; nt < 4; ++nt) ctx[nt][r] *= alpha[r];
    }

    // ---- stage P (f16) through LDS to re-shape D-frag -> A-frag --------
    _Float16* P = sP[wave];
#pragma unroll
    for (int nt = 0; nt < 4; ++nt)
#pragma unroll
      for (int r = 0; r < 8; ++r)
        P[(r + 8 * half) * 64 + nt * 16 + nl] = (_Float16)sc[nt][r];
    __syncthreads();

    // ---- ctx += P V  (key chunks of 32, B-fragments from LDS) ----------
#pragma unroll
    for (int c = 0; c < 2; ++c) {
      v16h ap = load_a_f16(P, 64, 0, c * 32, lane);
#pragma unroll
      for (int nt = 0; nt < 4; ++nt) {
        v16h bv_ = load_b_f16(Vtile, DKDIM, nt * 16, c * 32, lane);
        ctx[nt] = wmma_f16(ap, bv_, ctx[nt]);
      }
    }
    __syncthreads();  // everyone done with 'cur' before it is restaged
  }

  // ---- normalize and store ctx as f16 [B*S, EMB] -----------------------
#pragma unroll
  for (int r = 0; r < 8; ++r) {
    const float inv = 1.0f / lrow[r];
    const int s = q0 + r + 8 * half;
    const size_t row = (size_t)b * SEQ + s;
#pragma unroll
    for (int nt = 0; nt < 4; ++nt) {
      const int col = h * 64 + nt * 16 + nl;
      Ch[row * EMB + col] = (_Float16)(ctx[nt][r] * inv);
    }
  }
}

// ---------------------------------------------------------------------------
// Kernel 4: output projection  out = ctx @ Wf + bf  (f32 result)
// ---------------------------------------------------------------------------
__global__ __launch_bounds__(128) void final_proj(
    const _Float16* __restrict__ Ch, const _Float16* __restrict__ Wt,
    const float* __restrict__ bf, float* __restrict__ out) {
  const int lane = threadIdx.x & 31;
  const int wave = threadIdx.x >> 5;
  const int row0 = blockIdx.x * 64 + wave * 16;
  const int n0 = blockIdx.y * 64;

  v8f acc[4] = {};
  for (int k0 = 0; k0 < EMB; k0 += 32) {
    v16h a = load_a_f16(Ch, EMB, row0, k0, lane);
#pragma unroll
    for (int nt = 0; nt < 4; ++nt) {
      v16h bfrag = load_b_f16(Wt, EMB, n0 + nt * 16, k0, lane);
      acc[nt] = wmma_f16(a, bfrag, acc[nt]);
    }
  }

  const int half = lane >> 4, nl = lane & 15;
#pragma unroll
  for (int nt = 0; nt < 4; ++nt) {
    const int gc = n0 + nt * 16 + nl;
    const float bb = bf[gc];
#pragma unroll
    for (int r = 0; r < 8; ++r) {
      const int gr = row0 + r + 8 * half;
      out[(size_t)gr * EMB + gc] = acc[nt][r] + bb;
    }
  }
}

// ---------------------------------------------------------------------------
extern "C" void kernel_launch(void* const* d_in, const int* in_sizes, int n_in,
                              void* d_out, int out_size, void* d_ws,
                              size_t ws_size, hipStream_t stream) {
  (void)in_sizes; (void)n_in; (void)out_size; (void)ws_size;
  const float* query = (const float*)d_in[0];
  const float* key_  = (const float*)d_in[1];
  const float* value = (const float*)d_in[2];
  const int*   mask  = (const int*)d_in[3];
  const float* Wq = (const float*)d_in[4];
  const float* bq = (const float*)d_in[5];
  const float* Wk = (const float*)d_in[6];
  const float* bk = (const float*)d_in[7];
  const float* Wv = (const float*)d_in[8];
  const float* bv = (const float*)d_in[9];
  const float* Wf = (const float*)d_in[10];
  const float* bf = (const float*)d_in[11];
  float* out = (float*)d_out;

  // Workspace (f16 halves): Qh, Kh, Vt, Ch (6,291,456 each) + Wt[4] -- ~55 MB
  _Float16* ws = (_Float16*)d_ws;
  const size_t QSZ = (size_t)BATCH * HEADS * SEQ * DKDIM;
  _Float16* Qh = ws;
  _Float16* Kh = Qh + QSZ;
  _Float16* Vt = Kh + QSZ;
  _Float16* Ch = Vt + QSZ;
  _Float16* Wt = Ch + QSZ;

  {
    const int total = 4 * EMB * EMB;
    cvt_weights<<<total / 256, 256, 0, stream>>>(Wq, Wk, Wv, Wf, Wt);
  }
  qkv_proj<<<dim3(MROWS / 64, EMB / 64, 3), 128, 0, stream>>>(
      query, key_, value, Wt, bq, bk, bv, Qh, Kh, Vt);
  attn<<<dim3(SEQ / 128, HEADS, BATCH), 256, 0, stream>>>(Qh, Kh, Vt, mask, Ch);
  final_proj<<<dim3(MROWS / 64, EMB / 64), 128, 0, stream>>>(
      Ch, Wt + (size_t)3 * EMB * EMB, bf, out);
}
